// SparseFlashAttn_36687610643006
// MI455X (gfx1250) — compile-verified, matrix operand order checked by reading, not run on hardware
//
#include <hip/hip_runtime.h>
#include <hip/hip_bf16.h>

typedef __attribute__((ext_vector_type(16))) __bf16 v16bf;
typedef __attribute__((ext_vector_type(8)))  float  v8f;

#if defined(__has_builtin)
#if __has_builtin(__builtin_amdgcn_exp2f)
#define EXP2F(x) __builtin_amdgcn_exp2f(x)
#endif
#endif
#ifndef EXP2F
#define EXP2F(x) exp2f(x)
#endif

namespace {
constexpr int kB = 32, kH = 32, kHkv = 8, kD = 128, kDv = 128;
constexpr int kT = 4096, kS = 32, kBlockN = 64;
constexpr int kG = kH / kHkv;          // 4 (GQA group)
constexpr int kWaves = 8;              // waves per workgroup
constexpr int kPStride = 34;           // padded P-row stride (floats) -> no LDS bank conflicts
constexpr float kNegBig = -1e30f;
// softmax in exp2 domain: s * (1/sqrt(D)) * log2(e)
constexpr float kScaleLog2e = 0.08838834764831845f * 1.4426950408889634f;
}

__global__ __launch_bounds__(256, 1)
void sparse_flash_decode(const float* __restrict__ Q,
                         const float* __restrict__ K,
                         const float* __restrict__ V,
                         const int*   __restrict__ BI,
                         const int*   __restrict__ SEQ,
                         float* __restrict__ O)
{
  __shared__ float ldsP[kWaves][16][kPStride];   // per-wave P transpose scratch
  __shared__ float ldsAcc[kWaves][kG][kDv];      // per-wave partial outputs (rows 0..G-1)
  __shared__ float ldsM[kWaves][kG];
  __shared__ float ldsL[kWaves][kG];

  const int bh   = blockIdx.x;
  const int b    = bh / kHkv;
  const int kvh  = bh % kHkv;
  const int tid  = threadIdx.x;
  const int wave = tid >> 5;
  const int lane = tid & 31;
  const int l15  = lane & 15;
  const int hi   = lane >> 4;
  const int seqlen = SEQ[b];

  // ---- Q tile as bf16 A-fragments (16x32 each, 4 chunks cover D=128).
  // A layout: lane = M row; VGPR r holds K pair (r&3)*2 + (r>>2)*16 + hi*8.
  v16bf qf[4];
  {
    const bool qv = (l15 < kG);
    const float* qp = Q + ((size_t)b * kH + kvh * kG + (qv ? l15 : 0)) * kD;
#pragma unroll
    for (int c = 0; c < 4; ++c) {
#pragma unroll
      for (int r = 0; r < 8; ++r) {
        const int k = c * 32 + (r & 3) * 2 + (r >> 2) * 16 + hi * 8;
        const float2 v2 = *(const float2*)(qp + k);
        qf[c][2 * r]     = (__bf16)(qv ? v2.x : 0.0f);
        qf[c][2 * r + 1] = (__bf16)(qv ? v2.y : 0.0f);
      }
    }
  }

  v8f acc[8];
#pragma unroll
  for (int t = 0; t < 8; ++t)
#pragma unroll
    for (int j = 0; j < 8; ++j) acc[t][j] = 0.0f;

  float mrow[8], lrow[8];
#pragma unroll
  for (int j = 0; j < 8; ++j) { mrow[j] = kNegBig; lrow[j] = 0.0f; }

  const float* kbasep = K + ((size_t)b * kT * kHkv + kvh) * kD;
  const float* vbasep = V + ((size_t)b * kT * kHkv + kvh) * kDv + l15;

  // Each wave handles blocks wave, wave+8, wave+16, wave+24 (uniform trip count).
  for (int si = 0; si < kS / kWaves; ++si) {
    const int sblk = wave + si * kWaves;
    const int blk  = BI[((size_t)b * kHkv + kvh) * kS + sblk];
    const int tokbase = blk * kBlockN;

    for (int pair = 0; pair < 2; ++pair) {          // 32 tokens per pair
      // Precompute V row offsets (tokens for PV B-fragments), clamped in-bounds.
      int voff0[8], voff1[8];
#pragma unroll
      for (int r = 0; r < 8; ++r) {
        const int k = (r & 3) * 2 + (r >> 2) * 16 + hi * 8;
        int t0 = tokbase + pair * 32 + k;
        int t1 = t0 + 1;
        t0 = t0 < 0 ? 0 : (t0 > kT - 1 ? kT - 1 : t0);
        t1 = t1 < 0 ? 0 : (t1 > kT - 1 ? kT - 1 : t1);
        voff0[r] = t0 * (kHkv * kDv);
        voff1[r] = t1 * (kHkv * kDv);
      }

      // ---- QK^T: two 16x16 score tiles (f32 accum)
      v8f st[2];
#pragma unroll
      for (int half = 0; half < 2; ++half) {
        const int nt = pair * 2 + half;
        const int tok = tokbase + nt * 16 + l15;    // this lane's key column
        const int tokc = tok < 0 ? 0 : (tok > kT - 1 ? kT - 1 : tok);
        const float* kp = kbasep + (size_t)tokc * (kHkv * kD);
        v8f sv;
#pragma unroll
        for (int j = 0; j < 8; ++j) sv[j] = 0.0f;
#pragma unroll
        for (int c = 0; c < 4; ++c) {
          v16bf kb;  // B-frag 32x16: lane = token column, regs hold D-chunk pairs
#pragma unroll
          for (int r = 0; r < 8; ++r) {
            const int k = c * 32 + (r & 3) * 2 + (r >> 2) * 16 + hi * 8;
            const float2 v2 = *(const float2*)(kp + k);
            kb[2 * r]     = (__bf16)v2.x;
            kb[2 * r + 1] = (__bf16)v2.y;
          }
          sv = __builtin_amdgcn_wmma_f32_16x16x32_bf16(false, qf[c], false, kb,
                                                       (short)0, sv, false, false);
        }
        const bool valid = (blk >= 0) && (tok < seqlen);
#pragma unroll
        for (int j = 0; j < 8; ++j)
          sv[j] = valid ? sv[j] * kScaleLog2e : kNegBig;
        st[half] = sv;
      }

      // ---- online softmax update, per row M = j + 8*hi (reduce over 16-lane half)
#pragma unroll
      for (int j = 0; j < 8; ++j) {
        float v = fmaxf(st[0][j], st[1][j]);
        v = fmaxf(v, __shfl_xor(v, 1));
        v = fmaxf(v, __shfl_xor(v, 2));
        v = fmaxf(v, __shfl_xor(v, 4));
        v = fmaxf(v, __shfl_xor(v, 8));
        const float mn    = fmaxf(mrow[j], v);
        const float alpha = EXP2F(mrow[j] - mn);
        mrow[j] = mn;
        const float p0 = EXP2F(st[0][j] - mn);
        const float p1 = EXP2F(st[1][j] - mn);
        st[0][j] = p0; st[1][j] = p1;
        float rs = p0 + p1;
        rs += __shfl_xor(rs, 1);
        rs += __shfl_xor(rs, 2);
        rs += __shfl_xor(rs, 4);
        rs += __shfl_xor(rs, 8);
        lrow[j] = lrow[j] * alpha + rs;
#pragma unroll
        for (int t = 0; t < 8; ++t) acc[t][j] *= alpha;
      }

      // ---- P: C-layout -> per-wave LDS scratch -> bf16 A-fragment (16x32)
      // Same-wave DS ops complete in order (ISA §7.3); block compiler motion.
#pragma unroll
      for (int half = 0; half < 2; ++half)
#pragma unroll
        for (int j = 0; j < 8; ++j)
          ldsP[wave][j + 8 * hi][half * 16 + l15] = st[half][j];
      __builtin_amdgcn_sched_barrier(0);
      v16bf pf;
#pragma unroll
      for (int r = 0; r < 8; ++r) {
        const int k = (r & 3) * 2 + (r >> 2) * 16 + hi * 8;
        const float2 v2 = *(const float2*)&ldsP[wave][l15][k];
        pf[2 * r]     = (__bf16)v2.x;
        pf[2 * r + 1] = (__bf16)v2.y;
      }
      __builtin_amdgcn_sched_barrier(0);

      // ---- PV: accumulate 8 Dv tiles (coalesced V loads: 16 lanes contiguous)
#pragma unroll
      for (int dt = 0; dt < 8; ++dt) {
        v16bf vb;
#pragma unroll
        for (int r = 0; r < 8; ++r) {
          const float x0 = vbasep[voff0[r] + dt * 16];
          const float x1 = vbasep[voff1[r] + dt * 16];
          vb[2 * r]     = (__bf16)x0;
          vb[2 * r + 1] = (__bf16)x1;
        }
        acc[dt] = __builtin_amdgcn_wmma_f32_16x16x32_bf16(false, pf, false, vb,
                                                          (short)0, acc[dt], false, false);
      }
    }
  }

  // ---- publish per-wave partials (only real rows 0..G-1 live in hi==0, j<G)
  if (hi == 0) {
#pragma unroll
    for (int j = 0; j < kG; ++j)
#pragma unroll
      for (int dt = 0; dt < 8; ++dt)
        ldsAcc[wave][j][dt * 16 + l15] = acc[dt][j];
    if (l15 == 0) {
#pragma unroll
      for (int j = 0; j < kG; ++j) { ldsM[wave][j] = mrow[j]; ldsL[wave][j] = lrow[j]; }
    }
  }
  __syncthreads();

  // ---- split-softmax combine across the 8 waves; 256 threads -> 512 outputs
  for (int e = tid; e < kG * kDv; e += 256) {
    const int g  = e / kDv;
    const int dv = e % kDv;
    float M = kNegBig;
#pragma unroll
    for (int w = 0; w < kWaves; ++w) M = fmaxf(M, ldsM[w][g]);
    float Ls = 0.0f, os = 0.0f;
#pragma unroll
    for (int w = 0; w < kWaves; ++w) {
      const float sc = EXP2F(ldsM[w][g] - M);
      Ls += ldsL[w][g] * sc;
      os += ldsAcc[w][g][dv] * sc;
    }
    // all-invalid kv head -> zeros (matches reference any_valid behavior)
    const float out = (M > -1e29f && Ls > 0.0f) ? os / Ls : 0.0f;
    O[((size_t)b * kH + kvh * kG + g) * kDv + dv] = out;
  }
}

extern "C" void kernel_launch(void* const* d_in, const int* in_sizes, int n_in,
                              void* d_out, int out_size, void* d_ws, size_t ws_size,
                              hipStream_t stream) {
  (void)in_sizes; (void)n_in; (void)d_ws; (void)ws_size; (void)out_size;
  const float* Q  = (const float*)d_in[0];
  const float* K  = (const float*)d_in[1];
  const float* V  = (const float*)d_in[2];
  const int*   BI = (const int*)d_in[3];
  const int*   SQ = (const int*)d_in[4];
  float*       O  = (float*)d_out;

  dim3 grid(kB * kHkv);   // one workgroup per (batch, kv-head)
  dim3 block(256);        // 8 wave32
  sparse_flash_decode<<<grid, block, 0, stream>>>(Q, K, V, BI, SQ, O);
}
